// ScoreAttention_16329465660067
// MI455X (gfx1250) — compile-verified, matrix operand order checked by reading, not run on hardware
//
#include <hip/hip_runtime.h>

typedef __attribute__((ext_vector_type(16))) _Float16 v16h;
typedef __attribute__((ext_vector_type(8)))  _Float16 v8h;
typedef __attribute__((ext_vector_type(8)))  float    v8f;

namespace {
constexpr int kB   = 2;
constexpr int kS   = 2048;
constexpr int kH   = 16;
constexpr int kDH  = 64;
constexpr int kDIn = 1024;
}

__device__ __forceinline__ v16h pack_frag(v8h lo, v8h hi) {
  v16h r;
#pragma unroll
  for (int i = 0; i < 8; ++i) { r[i] = lo[i]; r[i + 8] = hi[i]; }
  return r;
}

// Async copy of 16 bytes global -> LDS (per lane), tracked by ASYNCcnt.
// CDNA5 ISA / CDNA5 ISA (opcode 98).
__device__ __forceinline__ void async_load16(void* lds_dst, const void* gsrc) {
  unsigned lds_off = (unsigned)(unsigned long long)lds_dst;  // addr[31:0] = LDS offset
  unsigned long long ga = (unsigned long long)gsrc;
  asm volatile("global_load_async_to_lds_b128 %0, %1, off"
               :: "v"(lds_off), "v"(ga)
               : "memory");
}

__device__ __forceinline__ void wait_async0() {
  asm volatile("s_wait_asynccnt 0x0" ::: "memory");
}

// ---------------------------------------------------------------------------
// Kernel 1: structured projections.
//   kraw[h,c] = sum_v softmax(W_K[h,:])[v] * x[p, v*64+c]   (same for q,v)
//   q[h,c']   = sum_c W_pred[h,c,c'] * qraw[h,c]
// Outputs (f16): Kf, Qf row-major [BH, S, 64]; Vt transposed [BH, 64, S].
// ---------------------------------------------------------------------------
__global__ __launch_bounds__(256) void proj_kernel(
    const float* __restrict__ x,
    const float* __restrict__ WK, const float* __restrict__ WQ,
    const float* __restrict__ WV, const float* __restrict__ Wpred,
    _Float16* __restrict__ Kf, _Float16* __restrict__ Qf,
    _Float16* __restrict__ Vt) {
  __shared__ float xs[4 * kDIn];
  __shared__ float qs[4 * kH * kDH];
  __shared__ float probs[3 * kH * 16];

  const int tid = threadIdx.x;

  if (tid < 48) {
    const float* Wm = (tid < 16) ? WK : ((tid < 32) ? WQ : WV);
    const int h = tid & 15;
    float w[16];
    float mx = -1e30f;
#pragma unroll
    for (int v = 0; v < 16; ++v) { w[v] = Wm[h * 16 + v]; mx = fmaxf(mx, w[v]); }
    float s = 0.f;
#pragma unroll
    for (int v = 0; v < 16; ++v) { w[v] = __expf(w[v] - mx); s += w[v]; }
    const float inv = 1.0f / s;
#pragma unroll
    for (int v = 0; v < 16; ++v) probs[(tid >> 4) * 256 + h * 16 + v] = w[v] * inv;
  }

  const int pos0 = blockIdx.x * 4;  // flattened b*S+p
  {
    const float4* xg  = (const float4*)(x + (size_t)pos0 * kDIn);
    float4*       xs4 = (float4*)xs;
    for (int i = tid; i < 4 * kDIn / 4; i += 256) xs4[i] = xg[i];
  }
  __syncthreads();

  {
    const int pl = tid >> 6;
    const int c  = tid & 63;
    const int row = pos0 + pl;
    const int b = row / kS, p = row % kS;
    const float* xr = xs + pl * kDIn;
#pragma unroll 1
    for (int h = 0; h < kH; ++h) {
      const float* pk = &probs[0 * 256 + h * 16];
      const float* pq = &probs[1 * 256 + h * 16];
      const float* pv = &probs[2 * 256 + h * 16];
      float kc = 0.f, qc = 0.f, vc = 0.f;
#pragma unroll
      for (int v = 0; v < 16; ++v) {
        const float xv = xr[v * 64 + c];
        kc = fmaf(pk[v], xv, kc);
        qc = fmaf(pq[v], xv, qc);
        vc = fmaf(pv[v], xv, vc);
      }
      const size_t bh = (size_t)(b * kH + h);
      Kf[(bh * kS + p) * kDH + c] = (_Float16)kc;
      Vt[(bh * kDH + c) * kS + p] = (_Float16)vc;
      qs[(pl * kH + h) * kDH + c] = qc;
    }
  }
  __syncthreads();

  {
    const int combo = tid >> 2;
    const int pl = combo >> 4, h = combo & 15;
    const int cq = (tid & 3) * 16;
    const int row = pos0 + pl;
    const int b = row / kS, p = row % kS;
    const float* qr = qs + (pl * kH + h) * kDH;
    const float* Wp = Wpred + (size_t)h * kDH * kDH;
    const size_t bh = (size_t)(b * kH + h);
#pragma unroll 1
    for (int j = 0; j < 16; ++j) {
      const int cc = cq + j;
      float acc = 0.f;
#pragma unroll
      for (int a = 0; a < kDH; ++a) acc = fmaf(Wp[a * kDH + cc], qr[a], acc);
      Qf[(bh * kS + p) * kDH + cc] = (_Float16)acc;
    }
  }
}

// ---------------------------------------------------------------------------
// Kernel 2: causal flash attention per (b,h).
// 128 threads = 4 waves share a 64-row query super-tile. Each 32-key K/V
// block is staged ONCE into LDS with global_load_async_to_lds_b128
// (double-buffered, s_wait_asynccnt), then all 4 waves run
// v_wmma_f32_16x16x32_f16 from LDS fragments with per-row online softmax.
// ---------------------------------------------------------------------------
__global__ __launch_bounds__(128) void attn_kernel(
    const _Float16* __restrict__ Kf, const _Float16* __restrict__ Qf,
    const _Float16* __restrict__ Vt, float* __restrict__ out) {
  __shared__ alignas(16) _Float16 Ks[2][32 * 64];  // [key 0..31][dim 0..63]
  __shared__ alignas(16) _Float16 Vs[2][64 * 32];  // [dim 0..63][key 0..31]
  __shared__ alignas(16) _Float16 Pl[4][16 * 32];  // per-wave P tile

  const int tid  = threadIdx.x;
  const int lane = tid & 31;
  const int wave = tid >> 5;
  const int half = lane >> 4;
  const int lm   = lane & 15;
  const int bh = blockIdx.y;
  const int b = bh >> 4, h = bh & 15;
  const int qbase = blockIdx.x * 64;
  const int q0 = qbase + wave * 16;  // this wave's query tile base

  const _Float16* Kb = Kf + (size_t)bh * kS * kDH;
  const _Float16* Qb = Qf + (size_t)bh * kS * kDH;
  const _Float16* Vb = Vt + (size_t)bh * kDH * kS;

  // Q A-fragments (16 rows x 32 dims x 2 K-steps); per-lane: two contiguous
  // runs of 8 f16 at K = 8*half and 16+8*half (ISA 7.12.2).
  v16h aQ[2];
  {
    const _Float16* qrow = Qb + (size_t)(q0 + lm) * kDH;
#pragma unroll
    for (int t = 0; t < 2; ++t) {
      aQ[t] = pack_frag(*(const v8h*)(qrow + 32 * t + 8 * half),
                        *(const v8h*)(qrow + 32 * t + 16 + 8 * half));
    }
  }

  v8f o0 = {}, o1 = {}, o2 = {}, o3 = {};
  float mrow[8], lsum[8];
#pragma unroll
  for (int r = 0; r < 8; ++r) { mrow[r] = -1e30f; lsum[r] = 0.f; }

  _Float16* Pw = Pl[wave];
  const float scale = 0.125f;  // 1 / (sqrt(64) * temp)

  // Stage one 32-key block (K: 32x64 f16, V: 64x32 f16) into LDS buffer.
  // 256 16B chunks each; 128 threads x 2 chunks.
  auto stage = [&](int kb, int buf) {
#pragma unroll
    for (int it = 0; it < 2; ++it) {
      const int q = tid + it * 128;
      const int r = q >> 3, c = q & 7;          // key row, 16B chunk
      async_load16(&Ks[buf][r * 64 + c * 8],
                   Kb + (size_t)(kb + r) * kDH + c * 8);
    }
#pragma unroll
    for (int it = 0; it < 2; ++it) {
      const int q = tid + it * 128;
      const int d = q >> 2, c = q & 3;          // dim row, 16B chunk
      async_load16(&Vs[buf][d * 32 + c * 8],
                   Vb + (size_t)d * kS + kb + c * 8);
    }
  };

  const int nIter = (qbase + 64) >> 5;  // uniform across the block
  stage(0, 0);

  for (int i = 0; i < nIter; ++i) {
    const int kb = i << 5;
    const int buf = i & 1;
    wait_async0();        // staged data for this iteration has landed
    __syncthreads();      // ...and prior readers of buf are done
    if (i + 1 < nIter) stage(kb + 32, buf ^ 1);

    if (kb < q0 + 16) {   // wave-uniform: skip fully-masked key blocks
      // ---- S = Q @ K^T from LDS
      v8f s0 = {}, s1 = {};
#pragma unroll
      for (int t = 0; t < 2; ++t) {
        const _Float16* kr0 = &Ks[buf][(lm) * 64 + 32 * t];
        v16h bK0 = pack_frag(*(const v8h*)(kr0 + 8 * half),
                             *(const v8h*)(kr0 + 16 + 8 * half));
        s0 = __builtin_amdgcn_wmma_f32_16x16x32_f16(false, aQ[t], false, bK0,
                                                    (short)0, s0, false, false);
        const _Float16* kr1 = &Ks[buf][(16 + lm) * 64 + 32 * t];
        v16h bK1 = pack_frag(*(const v8h*)(kr1 + 8 * half),
                             *(const v8h*)(kr1 + 16 + 8 * half));
        s1 = __builtin_amdgcn_wmma_f32_16x16x32_f16(false, aQ[t], false, bK1,
                                                    (short)0, s1, false, false);
      }

      // ---- scale + causal mask + online softmax
      float p0[8], p1[8], rm[8];
#pragma unroll
      for (int r = 0; r < 8; ++r) {
        const int m = q0 + r + 8 * half;
        float a = s0[r] * scale;
        if (kb + lm > m) a = -1e30f;
        float c = s1[r] * scale;
        if (kb + 16 + lm > m) c = -1e30f;
        p0[r] = a; p1[r] = c;
        rm[r] = fmaxf(a, c);
      }
#pragma unroll
      for (int off = 1; off < 16; off <<= 1) {
#pragma unroll
        for (int r = 0; r < 8; ++r) rm[r] = fmaxf(rm[r], __shfl_xor(rm[r], off, 32));
      }
      float alpha[8], rs[8];
#pragma unroll
      for (int r = 0; r < 8; ++r) {
        const float mn = fmaxf(mrow[r], rm[r]);
        alpha[r] = __expf(mrow[r] - mn);
        mrow[r] = mn;
        p0[r] = __expf(p0[r] - mn);
        p1[r] = __expf(p1[r] - mn);
        rs[r] = p0[r] + p1[r];
      }
#pragma unroll
      for (int off = 1; off < 16; off <<= 1) {
#pragma unroll
        for (int r = 0; r < 8; ++r) rs[r] += __shfl_xor(rs[r], off, 32);
      }
#pragma unroll
      for (int r = 0; r < 8; ++r) {
        lsum[r] = lsum[r] * alpha[r] + rs[r];
        o0[r] *= alpha[r]; o1[r] *= alpha[r];
        o2[r] *= alpha[r]; o3[r] *= alpha[r];
      }

      // ---- repack P through the per-wave LDS tile into A-fragment form
#pragma unroll
      for (int r = 0; r < 8; ++r) {
        const int m = r + 8 * half;
        Pw[m * 32 + lm]      = (_Float16)p0[r];
        Pw[m * 32 + 16 + lm] = (_Float16)p1[r];
      }
      asm volatile("s_wait_dscnt 0x0" ::: "memory");  // wave-local LDS RAW

      v16h aP;
      {
        const _Float16* prow = Pw + lm * 32;
        aP = pack_frag(*(const v8h*)(prow + 8 * half),
                       *(const v8h*)(prow + 16 + 8 * half));
      }

      // ---- O += P @ V from LDS (V stored [dim][key])
#pragma unroll
      for (int nt = 0; nt < 4; ++nt) {
        const _Float16* vrow = &Vs[buf][(nt * 16 + lm) * 32];
        v16h bV = pack_frag(*(const v8h*)(vrow + 8 * half),
                            *(const v8h*)(vrow + 16 + 8 * half));
        v8f& o = (nt == 0) ? o0 : (nt == 1) ? o1 : (nt == 2) ? o2 : o3;
        o = __builtin_amdgcn_wmma_f32_16x16x32_f16(false, aP, false, bV,
                                                   (short)0, o, false, false);
      }
    }

    __syncthreads();  // everyone done reading buf before it is re-staged
  }

  // ---- epilogue: normalize and scatter to [B, S, H*64]
  float inv[8];
#pragma unroll
  for (int r = 0; r < 8; ++r) inv[r] = 1.0f / lsum[r];
  float* outBase = out + (size_t)(b * kS + q0) * (kH * kDH) + h * kDH;
#pragma unroll
  for (int r = 0; r < 8; ++r) {
    const int m = r + 8 * half;
    float* orow = outBase + (size_t)m * (kH * kDH);
    orow[0  + lm] = o0[r] * inv[r];
    orow[16 + lm] = o1[r] * inv[r];
    orow[32 + lm] = o2[r] * inv[r];
    orow[48 + lm] = o3[r] * inv[r];
  }
}

extern "C" void kernel_launch(void* const* d_in, const int* in_sizes, int n_in,
                              void* d_out, int out_size, void* d_ws, size_t ws_size,
                              hipStream_t stream) {
  const float* x     = (const float*)d_in[0];
  const float* WK    = (const float*)d_in[1];
  const float* WQ    = (const float*)d_in[2];
  const float* WV    = (const float*)d_in[3];
  const float* Wpred = (const float*)d_in[4];

  const size_t elems = (size_t)kB * kH * kS * kDH;  // 4M f16 = 8MB each
  _Float16* Kf = (_Float16*)d_ws;
  _Float16* Qf = Kf + elems;
  _Float16* Vt = Qf + elems;

  proj_kernel<<<dim3(kB * kS / 4), 256, 0, stream>>>(x, WK, WQ, WV, Wpred,
                                                     Kf, Qf, Vt);
  attn_kernel<<<dim3(kS / 64, kB * kH), 128, 0, stream>>>(Kf, Qf, Vt,
                                                          (float*)d_out);
}
